// ESTM_4243427688719
// MI455X (gfx1250) — matrix operand order repaired and checked by reading, not compiled, not run on hardware
//
#include <hip/hip_runtime.h>

// ---------------------------------------------------------------------------
// Types for CDNA5 WMMA (wave32): v_wmma_f32_16x16x32_bf16
// ---------------------------------------------------------------------------
typedef __attribute__((ext_vector_type(16))) __bf16 v16bf;
typedef __attribute__((ext_vector_type(8)))  float  v8f;
typedef __attribute__((ext_vector_type(4)))  unsigned int u32x4;

constexpr int Bn  = 4;
constexpr int Cn  = 64;
constexpr int CCn = 128;
constexpr int Hn  = 256;
constexpr int Wn  = 256;

union BF16x16 { v16bf v; u32x4 q[2]; };

__device__ __forceinline__ v8f wmma_bf16(v16bf a, v16bf b, v8f c) {
  return __builtin_amdgcn_wmma_f32_16x16x32_bf16(false, a, false, b, (short)0, c,
                                                 false, false);
}

// A-fragment (16x32 bf16, M=rows, K contiguous in memory) from f32 row-major W.
// lane%16 = row, lane/16 selects K-halves {0..7,16..23} vs {8..15,24..31}.
__device__ __forceinline__ v16bf load_a_f32(const float* W, int ldk,
                                            int row_base, int k_base, int lane) {
  const float* p = W + (size_t)(row_base + (lane & 15)) * ldk + k_base + ((lane >> 4) << 3);
  v16bf a;
#pragma unroll
  for (int i = 0; i < 8; ++i) { a[i] = (__bf16)p[i]; a[i + 8] = (__bf16)p[i + 16]; }
  return a;
}

// A-fragment from bf16 row-major (global or LDS). 16B-aligned vector loads.
__device__ __forceinline__ v16bf load_a_bf16(const __bf16* W, int ldk,
                                             int row_base, int k_base, int lane) {
  const __bf16* p = W + (size_t)(row_base + (lane & 15)) * ldk + k_base + ((lane >> 4) << 3);
  BF16x16 u;
  u.q[0] = *(const u32x4*)(p);
  u.q[1] = *(const u32x4*)(p + 16);
  return u.v;
}

// B-fragment (32x16 bf16, K x N): lane%16 = column, lane/16 selects K {0..7,16..23} vs {8..15,24..31}.
// Source must be "column-major with K contiguous" i.e. pixel-major activations.
__device__ __forceinline__ v16bf load_b_bf16(const __bf16* X, int ldk,
                                             int col_base, int k_base, int lane) {
  const __bf16* p = X + (size_t)(col_base + (lane & 15)) * ldk + k_base + ((lane >> 4) << 3);
  BF16x16 u;
  u.q[0] = *(const u32x4*)(p);
  u.q[1] = *(const u32x4*)(p + 16);
  return u.v;
}

// ---------------------------------------------------------------------------
// Kernel 0: repack ResB weights [oc][ic][ky][kx] f32 -> [ky*3+kx][oc][ic] bf16
// ---------------------------------------------------------------------------
__global__ void k_repack(const float* w1, const float* w2, __bf16* p1, __bf16* p2) {
  int i = blockIdx.x * blockDim.x + threadIdx.x;   // i = (kk*128 + oc)*32 + ic
  if (i >= 9 * 128 * 32) return;
  int ic = i & 31;
  int oc = (i >> 5) & 127;
  int kk = i >> 12;
  p1[i] = (__bf16)w1[(oc * 32 + ic) * 9 + kk];
  p2[i] = (__bf16)w2[(oc * 32 + ic) * 9 + kk];
}

// ---------------------------------------------------------------------------
// Kernel 1: per-pixel stage. For each side:
//   hs = lrelu(sc0@cond), scale = sc1@hs ; hh = lrelu(sh0@cond), shift = sh1@hh
//   sft = catfea*(scale+1)+shift  -> ws (bf16, pixel-major [pix][128])
//   x   = bv@catfea               -> d_out (f32, NCHW) and xT ws (bf16, [row][c][w])
// Block: 256 thr = 8 waves, 128 pixels along W; each wave owns 16 pixels.
// ---------------------------------------------------------------------------
__global__ void k_pointwise(const float* catL, const float* catR,
                            const float* condL, const float* condR,
                            const float* bv_w, const float* bv_b,
                            const float* sc0_w, const float* sc0_b,
                            const float* sc1_w, const float* sc1_b,
                            const float* sh0_w, const float* sh0_b,
                            const float* sh1_w, const float* sh1_b,
                            __bf16* sftL, __bf16* sftR,
                            __bf16* xTL, __bf16* xTR,
                            float* outp) {
  const int side = blockIdx.z;
  const int row  = blockIdx.y;           // b*H + h
  const int b = row >> 8, h = row & 255;
  const int x0 = blockIdx.x * 128;
  const float* cat  = side ? catR  : catL;
  const float* cond = side ? condR : condL;
  __bf16* sft = side ? sftR : sftL;
  __bf16* xT  = side ? xTR  : xTL;
  float* xo = outp + (size_t)side * Bn * Cn * Hn * Wn;

  extern __shared__ char smem_raw[];
  __bf16* s_cat  = (__bf16*)smem_raw;        // [128][144]
  __bf16* s_cond = s_cat  + 128 * 144;       // [128][80]
  __bf16* s_h    = s_cond + 128 * 80;        // [128][80]
  __bf16* s_sc   = s_h    + 128 * 80;        // [128][144]
  const int tid = threadIdx.x;

#pragma unroll 4
  for (int it = 0; it < 32; ++it) {          // cond: 64ch x 128pix
    int idx = it * 256 + tid;
    int c = idx >> 7, p = idx & 127;
    s_cond[p * 80 + c] = (__bf16)cond[(((size_t)b * 64 + c) * Hn + h) * Wn + x0 + p];
  }
#pragma unroll 4
  for (int it = 0; it < 64; ++it) {          // catfea: 128ch x 128pix
    int idx = it * 256 + tid;
    int c = idx >> 7, p = idx & 127;
    s_cat[p * 144 + c] = (__bf16)cat[(((size_t)b * 128 + c) * Hn + h) * Wn + x0 + p];
  }
  __syncthreads();

  const int lane = tid & 31, wv = tid >> 5;
  const int pb = wv * 16;
  const int hl = lane >> 4, nn = lane & 15;

  // ---- scale branch
#pragma unroll
  for (int mt = 0; mt < 4; ++mt) {
    v8f acc = {};
#pragma unroll
    for (int ks = 0; ks < 2; ++ks)
      acc = wmma_bf16(load_a_f32(sc0_w, 64, mt * 16, ks * 32, lane),
                      load_b_bf16(s_cond, 80, pb, ks * 32, lane), acc);
#pragma unroll
    for (int v = 0; v < 8; ++v) {
      int c = mt * 16 + v + 8 * hl;
      float t = acc[v] + sc0_b[c];
      t = t >= 0.f ? t : 0.1f * t;
      s_h[(pb + nn) * 80 + c] = (__bf16)t;
    }
  }
#pragma unroll
  for (int mt = 0; mt < 8; ++mt) {
    v8f acc = {};
#pragma unroll
    for (int ks = 0; ks < 2; ++ks)
      acc = wmma_bf16(load_a_f32(sc1_w, 64, mt * 16, ks * 32, lane),
                      load_b_bf16(s_h, 80, pb, ks * 32, lane), acc);
#pragma unroll
    for (int v = 0; v < 8; ++v) {
      int c = mt * 16 + v + 8 * hl;
      s_sc[(pb + nn) * 144 + c] = (__bf16)(acc[v] + sc1_b[c]);
    }
  }
  // ---- shift branch (+ combine into SFT output)
#pragma unroll
  for (int mt = 0; mt < 4; ++mt) {
    v8f acc = {};
#pragma unroll
    for (int ks = 0; ks < 2; ++ks)
      acc = wmma_bf16(load_a_f32(sh0_w, 64, mt * 16, ks * 32, lane),
                      load_b_bf16(s_cond, 80, pb, ks * 32, lane), acc);
#pragma unroll
    for (int v = 0; v < 8; ++v) {
      int c = mt * 16 + v + 8 * hl;
      float t = acc[v] + sh0_b[c];
      t = t >= 0.f ? t : 0.1f * t;
      s_h[(pb + nn) * 80 + c] = (__bf16)t;
    }
  }
#pragma unroll
  for (int mt = 0; mt < 8; ++mt) {
    v8f acc = {};
#pragma unroll
    for (int ks = 0; ks < 2; ++ks)
      acc = wmma_bf16(load_a_f32(sh1_w, 64, mt * 16, ks * 32, lane),
                      load_b_bf16(s_h, 80, pb, ks * 32, lane), acc);
    union { __bf16 hv[8]; u32x4 q; } tmp;
    int p = pb + nn;
#pragma unroll
    for (int v = 0; v < 8; ++v) {
      int c = mt * 16 + v + 8 * hl;
      float cv = (float)s_cat[p * 144 + c];
      float sv = (float)s_sc[p * 144 + c];
      tmp.hv[v] = (__bf16)(cv * (sv + 1.f) + (acc[v] + sh1_b[c]));
    }
    *(u32x4*)&sft[((size_t)row * Wn + x0 + p) * 128 + mt * 16 + 8 * hl] = tmp.q;
  }
  // ---- x = bv @ catfea  (writes f32 base into d_out + bf16 transposed copy)
#pragma unroll
  for (int mt = 0; mt < 4; ++mt) {
    v8f acc = {};
#pragma unroll
    for (int ks = 0; ks < 4; ++ks)
      acc = wmma_bf16(load_a_f32(bv_w, 128, mt * 16, ks * 32, lane),
                      load_b_bf16(s_cat, 144, pb, ks * 32, lane), acc);
    int p = x0 + pb + nn;
#pragma unroll
    for (int v = 0; v < 8; ++v) {
      int c = mt * 16 + v + 8 * hl;
      float val = acc[v] + bv_b[c];
      xo[(((size_t)b * 64 + c) * Hn + h) * Wn + p] = val;
      xT[((size_t)row * 64 + c) * Wn + p] = (__bf16)val;
    }
  }
}

// ---------------------------------------------------------------------------
// Kernels 2/3: ResB 3x3 grouped conv (groups=4, 32ch/group) as 9 shifted
// K=32 WMMA GEMMs. MODE 0: +bias, lrelu -> y1. MODE 1: +bias +skip -> res.
// Block: 256 thr / 8 waves, 64 pixels along W, 8 oc-tiles x 4 pixel-subtiles.
// ---------------------------------------------------------------------------
template <int MODE>
__global__ void k_conv3x3(const __bf16* srcL, const __bf16* srcR,
                          const __bf16* wp, const float* bias,
                          const __bf16* skipL, const __bf16* skipR,
                          __bf16* dstL, __bf16* dstR) {
  const int side = blockIdx.z;
  const int row  = blockIdx.y;
  const int b = row >> 8, h = row & 255;
  const int x0 = blockIdx.x * 64;
  const __bf16* src  = side ? srcR  : srcL;
  const __bf16* skip = side ? skipR : skipL;
  __bf16* dst = side ? dstR : dstL;

  extern __shared__ char smem_raw[];
  __bf16* s_in = (__bf16*)smem_raw;          // [3][66][144]
  const int tid = threadIdx.x;

  for (int r = 0; r < 3; ++r) {
    int y = h + r - 1;
    bool yin = (unsigned)y < (unsigned)Hn;
    for (int it = 0; it < 17; ++it) {        // 66 pix * 64 dword-pairs
      int idx = it * 256 + tid;
      if (idx >= 66 * 64) break;
      int p = idx >> 6, cp = idx & 63;
      int x = x0 - 1 + p;
      unsigned int val = 0u;
      if (yin && (unsigned)x < (unsigned)Wn)
        val = *(const unsigned int*)&src[(((size_t)b * Hn + y) * Wn + x) * 128 + cp * 2];
      *(unsigned int*)&s_in[(r * 66 + p) * 144 + cp * 2] = val;
    }
  }
  __syncthreads();

  const int lane = tid & 31, wv = tid >> 5;
  const int hl = lane >> 4, nn = lane & 15;
  for (int t = wv; t < 32; t += 8) {
    int oct = t & 7, pt = t >> 3;
    int g = oct >> 1;                        // group (32 in-ch each)
    v8f acc = {};
#pragma unroll
    for (int ky = 0; ky < 3; ++ky)
#pragma unroll
      for (int kx = 0; kx < 3; ++kx)
        acc = wmma_bf16(load_a_bf16(wp + (size_t)(ky * 3 + kx) * 128 * 32, 32, oct * 16, 0, lane),
                        load_b_bf16(s_in + (size_t)(ky * 66) * 144, 144,
                                    pt * 16 + kx, g * 32, lane),
                        acc);
    union { __bf16 hv[8]; u32x4 q; } tmp;
    int p = pt * 16 + nn;
    size_t base = (((size_t)b * Hn + h) * Wn + x0 + p) * 128 + oct * 16 + 8 * hl;
#pragma unroll
    for (int v = 0; v < 8; ++v) {
      int c = oct * 16 + v + 8 * hl;
      float val = acc[v] + bias[c];
      if (MODE == 0) val = val >= 0.f ? val : 0.1f * val;   // lrelu
      else           val += (float)skip[base + v];          // residual skip
      tmp.hv[v] = (__bf16)val;
    }
    *(u32x4*)&dst[base] = tmp.q;
  }
}

// ---------------------------------------------------------------------------
// Kernel 4: grouped 1x1 proj (groups=2): side0 Q=bq(res_left), side1 K=bs(res_right)
// ---------------------------------------------------------------------------
__global__ void k_proj(const __bf16* resL, const __bf16* resR,
                       const float* bq_w, const float* bq_b,
                       const float* bs_w, const float* bs_b,
                       __bf16* qbuf, __bf16* kbuf) {
  const int side = blockIdx.z;
  const __bf16* src = side ? resR : resL;
  const float* wgt = side ? bs_w : bq_w;
  const float* bia = side ? bs_b : bq_b;
  __bf16* dst = side ? kbuf : qbuf;
  const int row = blockIdx.y;
  const int x0 = blockIdx.x * 128;

  extern __shared__ char smem_raw[];
  __bf16* s_r = (__bf16*)smem_raw;           // [128][144]
  const int tid = threadIdx.x;
#pragma unroll 4
  for (int it = 0; it < 32; ++it) {
    int idx = it * 256 + tid;
    int p = idx >> 6, cp = idx & 63;
    *(unsigned int*)&s_r[p * 144 + cp * 2] =
        *(const unsigned int*)&src[((size_t)row * Wn + x0 + p) * 128 + cp * 2];
  }
  __syncthreads();

  const int lane = tid & 31, wv = tid >> 5, hl = lane >> 4, nn = lane & 15;
  const int pb = wv * 16;
#pragma unroll
  for (int g = 0; g < 2; ++g)
#pragma unroll
    for (int mt = 0; mt < 2; ++mt) {
      v8f acc = {};
#pragma unroll
      for (int ks = 0; ks < 2; ++ks)
        acc = wmma_bf16(load_a_f32(wgt, 64, g * 32 + mt * 16, ks * 32, lane),
                        load_b_bf16(s_r, 144, pb, g * 64 + ks * 32, lane), acc);
      union { __bf16 hv[8]; u32x4 q; } tmp;
#pragma unroll
      for (int v = 0; v < 8; ++v)
        tmp.hv[v] = (__bf16)(acc[v] + bia[g * 32 + mt * 16 + v + 8 * hl]);
      *(u32x4*)&dst[((size_t)row * Wn + x0 + pb + nn) * 64 + g * 32 + mt * 16 + 8 * hl] = tmp.q;
    }
}

// ---------------------------------------------------------------------------
// Kernel 5: per-row parallax attention. One (b,h) row per block, 8 waves,
// each wave handles 2 query tiles of 16. side0: out_left += softmax(QK^T) x_r,
// side1: out_right += softmax(KQ^T) x_l  (score symmetry).
// ---------------------------------------------------------------------------
__global__ void k_attn(const __bf16* qbuf, const __bf16* kbuf,
                       const __bf16* xTL, const __bf16* xTR, float* outp) {
  const int side = blockIdx.y;
  const int row  = blockIdx.x;
  const int b = row >> 8, h = row & 255;
  const __bf16* Qs = side ? kbuf : qbuf;
  const __bf16* Ks = side ? qbuf : kbuf;
  const __bf16* xT = side ? xTL : xTR;
  float* xo = outp + (size_t)side * Bn * Cn * Hn * Wn;

  extern __shared__ char smem_raw[];
  __bf16* s_k = (__bf16*)smem_raw;           // [256][80]  keys pixel-major
  __bf16* s_x = s_k + 256 * 80;              // [64][272]  x channel-major
  __bf16* s_p = s_x + 64 * 272;              // [8 waves][16][272] probs
  const int tid = threadIdx.x;

#pragma unroll 4
  for (int it = 0; it < 32; ++it) {
    int idx = it * 256 + tid;
    int p = idx >> 5, cp = idx & 31;
    *(unsigned int*)&s_k[p * 80 + cp * 2] =
        *(const unsigned int*)&Ks[((size_t)row * Wn + p) * 64 + cp * 2];
  }
#pragma unroll 4
  for (int it = 0; it < 32; ++it) {
    int idx = it * 256 + tid;
    int c = idx >> 7, wp = idx & 127;
    *(unsigned int*)&s_x[c * 272 + wp * 2] =
        *(const unsigned int*)&xT[((size_t)row * 64 + c) * Wn + wp * 2];
  }
  __syncthreads();

  const int lane = tid & 31, wv = tid >> 5, hl = lane >> 4, nn = lane & 15;
  __bf16* s_pw = s_p + (size_t)wv * 16 * 272;

  for (int qi = 0; qi < 2; ++qi) {
    int qt = wv + qi * 8;
    v16bf aq0 = load_a_bf16(Qs + (size_t)row * Wn * 64, 64, qt * 16, 0, lane);
    v16bf aq1 = load_a_bf16(Qs + (size_t)row * Wn * 64, 64, qt * 16, 32, lane);
    v8f acc[16];
#pragma unroll
    for (int kt = 0; kt < 16; ++kt) {
      v8f c = {};
      c = wmma_bf16(aq0, load_b_bf16(s_k, 80, kt * 16, 0, lane), c);
      c = wmma_bf16(aq1, load_b_bf16(s_k, 80, kt * 16, 32, lane), c);
      acc[kt] = c;
    }
    // row-wise softmax: rows live on (VGPR v, lane-half); columns on lanes 0..15
    v8f mx = acc[0];
#pragma unroll
    for (int kt = 1; kt < 16; ++kt)
#pragma unroll
      for (int v = 0; v < 8; ++v) mx[v] = fmaxf(mx[v], acc[kt][v]);
#pragma unroll
    for (int m = 1; m <= 8; m <<= 1)
#pragma unroll
      for (int v = 0; v < 8; ++v) mx[v] = fmaxf(mx[v], __shfl_xor(mx[v], m, 32));
    v8f sm = {};
#pragma unroll
    for (int kt = 0; kt < 16; ++kt)
#pragma unroll
      for (int v = 0; v < 8; ++v) {
        float p = __expf(acc[kt][v] - mx[v]);
        sm[v] += p;
        s_pw[(v + 8 * hl) * 272 + kt * 16 + nn] = (__bf16)p;
      }
#pragma unroll
    for (int m = 1; m <= 8; m <<= 1)
#pragma unroll
      for (int v = 0; v < 8; ++v) sm[v] += __shfl_xor(sm[v], m, 32);
    // out(16 queries x 64 ch) = P (16x256) @ x (256x64), normalized by row sums
#pragma unroll
    for (int ct = 0; ct < 4; ++ct) {
      v8f o = {};
#pragma unroll
      for (int ks = 0; ks < 8; ++ks)
        o = wmma_bf16(load_a_bf16(s_pw, 272, 0, ks * 32, lane),
                      load_b_bf16(s_x, 272, ct * 16, ks * 32, lane), o);
      int c = ct * 16 + nn;
#pragma unroll
      for (int v = 0; v < 8; ++v) {
        int q = qt * 16 + v + 8 * hl;
        size_t addr = (((size_t)b * 64 + c) * Hn + h) * Wn + q;
        xo[addr] = xo[addr] + o[v] / sm[v];   // x + xT (x written by k_pointwise)
      }
    }
  }
}

// ---------------------------------------------------------------------------
extern "C" void kernel_launch(void* const* d_in, const int* in_sizes, int n_in,
                              void* d_out, int out_size, void* d_ws, size_t ws_size,
                              hipStream_t stream) {
  const float* catL  = (const float*)d_in[0];
  const float* catR  = (const float*)d_in[1];
  const float* condL = (const float*)d_in[2];
  const float* condR = (const float*)d_in[3];
  const float* bv_w  = (const float*)d_in[4];
  const float* bv_b  = (const float*)d_in[5];
  const float* bq_w  = (const float*)d_in[6];
  const float* bq_b  = (const float*)d_in[7];
  const float* bs_w  = (const float*)d_in[8];
  const float* bs_b  = (const float*)d_in[9];
  const float* rb_w1 = (const float*)d_in[10];
  const float* rb_b1 = (const float*)d_in[11];
  const float* rb_w2 = (const float*)d_in[12];
  const float* rb_b2 = (const float*)d_in[13];
  const float* sc0_w = (const float*)d_in[14];
  const float* sc0_b = (const float*)d_in[15];
  const float* sc1_w = (const float*)d_in[16];
  const float* sc1_b = (const float*)d_in[17];
  const float* sh0_w = (const float*)d_in[18];
  const float* sh0_b = (const float*)d_in[19];
  const float* sh1_w = (const float*)d_in[20];
  const float* sh1_b = (const float*)d_in[21];
  float* outp = (float*)d_out;

  const size_t PIX = (size_t)Bn * Hn * Wn;   // 262144
  char* ws = (char*)d_ws;
  size_t off = 0;
  auto carve = [&](size_t bytes) { char* p = ws + off; off += (bytes + 255) & ~(size_t)255; return p; };
  __bf16* sftL = (__bf16*)carve(PIX * 128 * 2);
  __bf16* sftR = (__bf16*)carve(PIX * 128 * 2);
  __bf16* y1L  = (__bf16*)carve(PIX * 128 * 2);
  __bf16* y1R  = (__bf16*)carve(PIX * 128 * 2);
  __bf16* xTL  = (__bf16*)carve(PIX * 64 * 2);
  __bf16* xTR  = (__bf16*)carve(PIX * 64 * 2);
  __bf16* qb   = (__bf16*)carve(PIX * 64 * 2);
  __bf16* kb   = (__bf16*)carve(PIX * 64 * 2);
  __bf16* rp1  = (__bf16*)carve((size_t)9 * 128 * 32 * 2);
  __bf16* rp2  = (__bf16*)carve((size_t)9 * 128 * 32 * 2);

  // 0) repack ResB weights for contiguous A-fragments
  k_repack<<<144, 256, 0, stream>>>(rb_w1, rb_w2, rp1, rp2);

  // 1) SFT + bv (both sides)
  {
    dim3 g(Wn / 128, Bn * Hn, 2);
    size_t lds = (size_t)(128 * 144 + 128 * 80 + 128 * 80 + 128 * 144) * 2; // 112KB
    k_pointwise<<<g, 256, lds, stream>>>(catL, catR, condL, condR,
                                         bv_w, bv_b, sc0_w, sc0_b, sc1_w, sc1_b,
                                         sh0_w, sh0_b, sh1_w, sh1_b,
                                         sftL, sftR, xTL, xTR, outp);
  }
  // 2) ResB conv1 (+lrelu) ; 3) ResB conv2 (+skip, in-place into sft buffer)
  {
    dim3 g(Wn / 64, Bn * Hn, 2);
    size_t lds = (size_t)3 * 66 * 144 * 2;   // ~55.7KB
    k_conv3x3<0><<<g, 256, lds, stream>>>(sftL, sftR, rp1, rb_b1, sftL, sftR, y1L, y1R);
    k_conv3x3<1><<<g, 256, lds, stream>>>(y1L, y1R, rp2, rb_b2, sftL, sftR, sftL, sftR);
  }
  // 4) grouped 1x1 projections -> Q (left), K (right)
  {
    dim3 g(Wn / 128, Bn * Hn, 2);
    size_t lds = (size_t)128 * 144 * 2;      // 36KB
    k_proj<<<g, 256, lds, stream>>>(sftL, sftR, bq_w, bq_b, bs_w, bs_b, qb, kb);
  }
  // 5) per-row parallax attention, both directions
  {
    dim3 g(Bn * Hn, 2);
    size_t lds = (size_t)(256 * 80 + 64 * 272 + 8 * 16 * 272) * 2; // ~142KB
    k_attn<<<g, 256, lds, stream>>>(qb, kb, xTL, xTR, outp);
  }
  (void)in_sizes; (void)n_in; (void)out_size; (void)ws_size;
}
